// ProteinFeatures_71657234367154
// MI455X (gfx1250) — matrix operand order repaired and checked by reading, not compiled, not run on hardware
//
#include <hip/hip_runtime.h>
#include <hip/hip_bf16.h>

typedef __attribute__((ext_vector_type(16))) _Float16 v16h;
typedef __attribute__((ext_vector_type(8)))  float    v8f;

#define NB    4
#define LSEQ  2048
#define TOPK  30
#define NFEAT 416
#define NOUT  128
#define KSTEPS 13          // 416 / 32
#define NEDGE (NB * LSEQ * TOPK)   // 245760
#define BLRES (NB * LSEQ)          // 8192

__constant__ int c_pair_i[24] = {0,2,3,4,1,1,1,1,0,0,0,4,4,3,0,2,3,4,2,3,4,2,3,2};
__constant__ int c_pair_j[24] = {0,2,3,4,0,2,3,4,2,3,4,2,3,2,1,1,1,1,0,0,0,4,4,3};

// ---------------------------------------------------------------------------
// Kernel 1: atom stack [N, Ca, C, O, Cb] -> ws  (layout atoms[5][B*L][3])
// ---------------------------------------------------------------------------
__global__ __launch_bounds__(256) void prep_atoms_kernel(const float* __restrict__ X,
                                                         float* __restrict__ atoms) {
    int idx = blockIdx.x * 256 + threadIdx.x;
    if (idx >= BLRES) return;
    const float* x = X + (long)idx * 15;   // 5 atoms x 3
    float N[3], Ca[3], C[3], O[3], Cb[3];
    #pragma unroll
    for (int d = 0; d < 3; ++d) { N[d] = x[d]; Ca[d] = x[3+d]; C[d] = x[6+d]; O[d] = x[12+d]; }
    float bv[3], cv[3], av[3];
    #pragma unroll
    for (int d = 0; d < 3; ++d) { bv[d] = Ca[d] - N[d]; cv[d] = C[d] - Ca[d]; }
    av[0] = bv[1]*cv[2] - bv[2]*cv[1];
    av[1] = bv[2]*cv[0] - bv[0]*cv[2];
    av[2] = bv[0]*cv[1] - bv[1]*cv[0];
    #pragma unroll
    for (int d = 0; d < 3; ++d)
        Cb[d] = -0.58273431f*av[d] + 0.56802827f*bv[d] - 0.54067466f*cv[d] + Ca[d];
    long base = (long)idx * 3;
    #pragma unroll
    for (int d = 0; d < 3; ++d) {
        atoms[0L*BLRES*3 + base + d] = N[d];
        atoms[1L*BLRES*3 + base + d] = Ca[d];
        atoms[2L*BLRES*3 + base + d] = C[d];
        atoms[3L*BLRES*3 + base + d] = O[d];
        atoms[4L*BLRES*3 + base + d] = Cb[d];
    }
}

// ---------------------------------------------------------------------------
// Kernel 2: pack edge_W (128x416 f32) into f16 WMMA-B fragment order:
//   bpack[((nt*13 + kk)*32 + lane)*8 + v] = {W[n][f], W[n][f+1]}
//   n = nt*16 + lane%16 ; k_local = (lane/16)*16 + 2v ; f = kk*32 + k_local
// ---------------------------------------------------------------------------
__global__ __launch_bounds__(256) void pack_weights_kernel(const float* __restrict__ W,
                                                           unsigned int* __restrict__ bpack) {
    int i = blockIdx.x * 256 + threadIdx.x;
    const int total = 8 * KSTEPS * 32 * 8;    // 26624
    if (i >= total) return;
    int v    = i & 7;
    int lane = (i >> 3) & 31;
    int t    = i >> 8;
    int kk   = t % KSTEPS;
    int nt   = t / KSTEPS;
    int n  = nt * 16 + (lane & 15);
    int h  = lane >> 4;
    int f  = kk * 32 + h * 16 + v * 2;
    union { _Float16 hh[2]; unsigned int u; } pk;
    pk.hh[0] = (_Float16)W[(long)n * NFEAT + f];
    pk.hh[1] = (_Float16)W[(long)n * NFEAT + f + 1];
    bpack[i] = pk.u;
}

// ---------------------------------------------------------------------------
// Kernel 3: masked top-K per row. One 256-thread block per (b,i).
// Writes E_idx (int, ws), D_neighbors (f32, ws), float(E_idx) (d_out tail).
// ---------------------------------------------------------------------------
__global__ __launch_bounds__(256) void topk_kernel(const float* __restrict__ atoms,
                                                   const float* __restrict__ maskp,
                                                   int* __restrict__ eidx_ws,
                                                   float* __restrict__ dnb_ws,
                                                   float* __restrict__ out_idx) {
    __shared__ float dist[LSEQ];
    __shared__ unsigned long long red[8];
    __shared__ float fred[8];
    __shared__ float sDmax;

    int row  = blockIdx.x;          // b*L + i
    int bb   = row / LSEQ;
    int tid  = threadIdx.x;
    int lane = tid & 31, wid = tid >> 5;

    const float* caBase = atoms + 1L * BLRES * 3;          // Ca plane
    const float* cai = caBase + (long)row * 3;
    float cx = cai[0], cy = cai[1], cz = cai[2];
    float mi = maskp[row];
    const float* caB = caBase + (long)bb * LSEQ * 3;
    const float* mB  = maskp + (long)bb * LSEQ;

    float lmax = 0.0f;
    for (int j = tid; j < LSEQ; j += 256) {
        const float* cj = caB + (long)j * 3;
        float dx = cx - cj[0], dy = cy - cj[1], dz = cz - cj[2];
        float raw = sqrtf(dx*dx + dy*dy + dz*dz + 1e-6f);
        float D = mi * mB[j] * raw;
        dist[j] = D;
        lmax = fmaxf(lmax, D);
    }
    #pragma unroll
    for (int o = 16; o >= 1; o >>= 1) lmax = fmaxf(lmax, __shfl_xor(lmax, o, 32));
    if (lane == 0) fred[wid] = lmax;
    __syncthreads();
    if (tid == 0) {
        float m = fred[0];
        #pragma unroll
        for (int w = 1; w < 8; ++w) m = fmaxf(m, fred[w]);
        sDmax = m;
    }
    __syncthreads();
    float Dmax = sDmax;
    for (int j = tid; j < LSEQ; j += 256)
        dist[j] += (1.0f - mi * mB[j]) * Dmax;
    __syncthreads();

    // 30 rounds of argmin with lowest-index tie-break
    for (int k = 0; k < TOPK; ++k) {
        unsigned long long key = ~0ull;
        #pragma unroll
        for (int it = 0; it < LSEQ / 256; ++it) {
            int j = tid + it * 256;
            unsigned long long cand =
                ((unsigned long long)__float_as_uint(dist[j]) << 32) | (unsigned int)j;
            key = (cand < key) ? cand : key;
        }
        #pragma unroll
        for (int o = 16; o >= 1; o >>= 1) {
            unsigned int hi = __shfl_xor((unsigned int)(key >> 32), o, 32);
            unsigned int lo = __shfl_xor((unsigned int)(key & 0xffffffffu), o, 32);
            unsigned long long other = ((unsigned long long)hi << 32) | lo;
            key = (other < key) ? other : key;
        }
        if (lane == 0) red[wid] = key;
        __syncthreads();
        if (tid == 0) {
            unsigned long long best = red[0];
            #pragma unroll
            for (int w = 1; w < 8; ++w) best = (red[w] < best) ? red[w] : best;
            int   j   = (int)(best & 0xffffffffu);
            float val = __uint_as_float((unsigned int)(best >> 32));
            long eo = (long)row * TOPK + k;
            eidx_ws[eo] = j;
            dnb_ws[eo]  = val;
            out_idx[eo] = (float)j;
            dist[j] = 3.0e38f;
        }
        __syncthreads();
    }
}

// ---------------------------------------------------------------------------
// Kernel 4: fused feature build + f16 WMMA GEMM (16x128 per block) + LayerNorm
// ---------------------------------------------------------------------------
__global__ __launch_bounds__(256) void edge_fused_kernel(
        const float* __restrict__ atoms,
        const int*   __restrict__ eidx,
        const float* __restrict__ dnb,
        const int*   __restrict__ ridx,
        const int*   __restrict__ chain,
        const float* __restrict__ posW,
        const float* __restrict__ posb,
        const unsigned int* __restrict__ bpack,
        const float* __restrict__ ln_g,
        const float* __restrict__ ln_b,
        float* __restrict__ out) {
    __shared__ __align__(16) _Float16 feat[16][NFEAT];   // 13,312 B
    __shared__ float othr[16 * NOUT];                    //  8,192 B
    __shared__ int   sJ[16], sD[16], sB[16], sI[16];
    __shared__ float sDn[16];

    int tid = threadIdx.x;
    long e0 = (long)blockIdx.x * 16;

    // phase 0: per-edge metadata
    if (tid < 16) {
        long e = e0 + tid;
        long r = e / TOPK;
        int ii = (int)(r % LSEQ);
        int bb = (int)(r / LSEQ);
        int jj = eidx[e];
        sJ[tid] = jj; sB[tid] = bb; sI[tid] = ii;
        sDn[tid] = dnb[e];
        int off  = ridx[bb * LSEQ + ii] - ridx[bb * LSEQ + jj];
        int same = (chain[bb * LSEQ + ii] == chain[bb * LSEQ + jj]);
        int d = off + 32;
        d = d < 0 ? 0 : (d > 64 ? 64 : d);
        sD[tid] = same ? d : 65;
    }
    __syncthreads();

    // phase A: 25 distances x 16 RBF per edge -> feat[e][16 .. 416)
    for (int idx = tid; idx < 16 * 25; idx += 256) {
        int el = idx / 25, p = idx % 25;
        float D;
        if (p == 0) {
            D = sDn[el];
        } else {
            int pi = c_pair_i[p - 1], pj = c_pair_j[p - 1];
            int bb = sB[el];
            const float* Ai = atoms + ((long)pi * BLRES + bb * LSEQ + sI[el]) * 3;
            const float* Aj = atoms + ((long)pj * BLRES + bb * LSEQ + sJ[el]) * 3;
            float dx = Ai[0]-Aj[0], dy = Ai[1]-Aj[1], dz = Ai[2]-Aj[2];
            D = sqrtf(dx*dx + dy*dy + dz*dz + 1e-6f);
        }
        union { _Float16 hh[16]; uint4 q[2]; } rb;
        #pragma unroll
        for (int r = 0; r < 16; ++r) {
            float mu = 2.0f + (20.0f / 15.0f) * (float)r;
            float t = (D - mu) * 0.8f;          // sigma = 1.25
            rb.hh[r] = (_Float16)__expf(-t * t);
        }
        uint4* dst = (uint4*)&feat[el][16 + p * 16];
        dst[0] = rb.q[0];
        dst[1] = rb.q[1];
    }
    // phase B: positional encoding -> feat[e][0..16)
    {
        int el = tid >> 4, c = tid & 15;                 // exactly 256 work items
        feat[el][c] = (_Float16)(posW[c * 66 + sD[el]] + posb[c]);
    }
    __syncthreads();

    // GEMM: wave w owns output columns [16w, 16w+16)
    int wid = tid >> 5, lane = tid & 31;
    int m = lane & 15, h = lane >> 4;
    union { v16h h16; unsigned int u[8]; uint4 q[2]; } A, Bm;
    v8f acc = {};
    const unsigned int* bp = bpack + (long)wid * KSTEPS * 256 + lane * 8;
    for (int kk = 0; kk < KSTEPS; ++kk) {
        // A fragment: two aligned b128 LDS loads (16-bit A 16x32 striping)
        A.q[0] = *(const uint4*)&feat[m][kk * 32 + h * 16];
        A.q[1] = *(const uint4*)&feat[m][kk * 32 + h * 16 + 8];
        // B fragment: two coalesced b128 global loads (L2-resident weights)
        const uint4* bp4 = (const uint4*)(bp + (long)kk * 256);
        Bm.q[0] = bp4[0];
        Bm.q[1] = bp4[1];
        acc = __builtin_amdgcn_wmma_f32_16x16x32_f16(
            false, A.h16, false, Bm.h16, (short)0, acc, false, false);
    }
    // scatter accumulator to LDS [16 edges][128 cols]
    #pragma unroll
    for (int v = 0; v < 8; ++v)
        othr[(v + 8 * h) * NOUT + wid * 16 + m] = acc[v];
    __syncthreads();

    // LayerNorm over 128 cols: 16 threads per edge, 8 cols each
    int el   = tid >> 4;
    int part = tid & 15;
    const float* rowp = &othr[el * NOUT + part * 8];
    float vals[8], s = 0.0f, ss = 0.0f;
    #pragma unroll
    for (int q = 0; q < 8; ++q) { vals[q] = rowp[q]; s += vals[q]; ss += vals[q] * vals[q]; }
    #pragma unroll
    for (int o = 8; o >= 1; o >>= 1) { s += __shfl_xor(s, o, 16); ss += __shfl_xor(ss, o, 16); }
    float mean = s * (1.0f / 128.0f);
    float var  = ss * (1.0f / 128.0f) - mean * mean;
    float rstd = rsqrtf(var + 1e-5f);
    int cb = part * 8;
    float ov[8];
    #pragma unroll
    for (int q = 0; q < 8; ++q)
        ov[q] = (vals[q] - mean) * rstd * ln_g[cb + q] + ln_b[cb + q];
    float* op = out + (e0 + el) * NOUT + cb;
    ((float4*)op)[0] = make_float4(ov[0], ov[1], ov[2], ov[3]);
    ((float4*)op)[1] = make_float4(ov[4], ov[5], ov[6], ov[7]);
}

// ---------------------------------------------------------------------------
extern "C" void kernel_launch(void* const* d_in, const int* in_sizes, int n_in,
                              void* d_out, int out_size, void* d_ws, size_t ws_size,
                              hipStream_t stream) {
    const float* X      = (const float*)d_in[0];
    const float* mask   = (const float*)d_in[1];
    const int*   ridx   = (const int*)  d_in[2];
    const int*   chain  = (const int*)  d_in[3];
    const float* posW   = (const float*)d_in[4];
    const float* posb   = (const float*)d_in[5];
    const float* edgeW  = (const float*)d_in[6];
    const float* ln_g   = (const float*)d_in[7];
    const float* ln_b   = (const float*)d_in[8];
    float* out = (float*)d_out;

    // workspace carve-up
    char* ws = (char*)d_ws;
    float* atoms        = (float*)ws;                              // 491,520 B
    int*   eidx_ws      = (int*)  (ws + 491520);                   // 983,040 B
    float* dnb_ws       = (float*)(ws + 491520 + 983040);          // 983,040 B
    unsigned int* bpack = (unsigned int*)(ws + 491520 + 2*983040); // 106,496 B

    float* out_idx = out + (long)NEDGE * NOUT;   // E_idx (as float) after E

    prep_atoms_kernel<<<(BLRES + 255) / 256, 256, 0, stream>>>(X, atoms);
    pack_weights_kernel<<<(8 * KSTEPS * 32 * 8 + 255) / 256, 256, 0, stream>>>(edgeW, bpack);
    topk_kernel<<<BLRES, 256, 0, stream>>>(atoms, mask, eidx_ws, dnb_ws, out_idx);
    edge_fused_kernel<<<NEDGE / 16, 256, 0, stream>>>(
        atoms, eidx_ws, dnb_ws, ridx, chain, posW, posb, bpack, ln_g, ln_b, out);
}
